// CausalMultiHeadAttentionLayer_76038101008907
// MI455X (gfx1250) — matrix operand order repaired and checked by reading, not compiled, not run on hardware
//
#include <hip/hip_runtime.h>

// ---------------------------------------------------------------------------
// Causal MHA for MI455X (gfx1250, wave32): WMMA bf16 + TDM-staged GEMMs
// ---------------------------------------------------------------------------

typedef __bf16 bf16_t;
typedef __attribute__((ext_vector_type(16))) __bf16 v16bf;
typedef __attribute__((ext_vector_type(8)))  __bf16 v8bf;
typedef __attribute__((ext_vector_type(8)))  float  v8f;
typedef __attribute__((ext_vector_type(4)))  unsigned int u32x4;
typedef __attribute__((ext_vector_type(8)))  int i32x8;
typedef __attribute__((ext_vector_type(4)))  int i32x4;

#define B_    4
#define S_    2048
#define D_    1024
#define H_    16
#define DH_   64
#define M_    (B_ * S_)     // 8192 tokens
#define NQKV_ (3 * D_)      // 3072

// LDS weight panel: 64 n-rows x 64 k, TDM-padded to 36 dwords (72 bf16) per row
#define PANEL_LDW 72
#define PANEL_ELEMS (64 * PANEL_LDW)

#define WMMA_BF16(a, b, c) \
    __builtin_amdgcn_wmma_f32_16x16x32_bf16(false, (a), false, (b), (short)0, (c), false, false)

// A-fragment (16x32 bf16 MxK), contiguous k = 16*lh + s. (global memory)
static __device__ __forceinline__ v16bf
load_a_frag(const bf16_t* base, int ld, int row, int kc, int lh) {
    const v8bf* p0 = reinterpret_cast<const v8bf*>(base + (long)row * ld + kc + 16 * lh);
    const v8bf* p1 = reinterpret_cast<const v8bf*>(base + (long)row * ld + kc + 16 * lh + 8);
    v8bf lo = *p0, hi = *p1;
    return __builtin_shufflevector(lo, hi, 0, 1, 2, 3, 4, 5, 6, 7, 8, 9, 10, 11, 12, 13, 14, 15);
}

// B-fragment (32x16 KxN) from a [N,K]-row-major GLOBAL buffer.
static __device__ __forceinline__ v16bf
load_b_frag(const bf16_t* baseT, int ld, int col, int kc, int lh) {
    const v8bf* p0 = reinterpret_cast<const v8bf*>(baseT + (long)col * ld + kc + 16 * lh);
    const v8bf* p1 = reinterpret_cast<const v8bf*>(baseT + (long)col * ld + kc + 16 * lh + 8);
    v8bf lo = *p0, hi = *p1;
    return __builtin_shufflevector(lo, hi, 0, 1, 2, 3, 4, 5, 6, 7, 8, 9, 10, 11, 12, 13, 14, 15);
}

// ---------------------------------------------------------------------------
// Tensor Data Mover: async-load a 64(n) x 64(k) bf16 tile of WT[N,K] into LDS,
// padding each 128B row with 16B (bank-conflict-free, 16B-aligned rows).
// D# per CDNA5 ISA.
// ---------------------------------------------------------------------------
static __device__ __forceinline__ void
tdm_load_panel(const bf16_t* tile_src, unsigned lds_byte_addr, int Ktot, int Ntot) {
    unsigned long long ga = (unsigned long long)tile_src;
    u32x4 g0;
    g0.x = 0x1u;                                   // count=1, user descriptor
    g0.y = lds_byte_addr;                          // lds_addr
    g0.z = (unsigned)(ga & 0xffffffffu);           // global_addr[31:0]
    g0.w = (unsigned)((ga >> 32) & 0x1ffffffu)     // global_addr[56:32]
         | (2u << 30);                             // type = 2 ("image")
    i32x8 g1;
    g1[0] = (1 << 16)                              // data_size = 1 (2 bytes)
          | (1 << 20)                              // pad_enable
          | (4 << 22)                              // pad_interval: 32 dwords
          | (3 << 25);                             // pad_amount: 4 dwords (16B)
    g1[1] = (Ktot & 0xffff) << 16;                 // tensor_dim0[15:0]
    g1[2] = ((Ktot >> 16) & 0xffff)                // tensor_dim0[31:16]
          | ((Ntot & 0xffff) << 16);               // tensor_dim1[15:0]
    g1[3] = ((Ntot >> 16) & 0xffff)                // tensor_dim1[31:16]
          | (64 << 16);                            // tile_dim0 = 64 (k)
    g1[4] = 64;                                    // tile_dim1 = 64 (n), tile_dim2 = 0
    g1[5] = Ktot;                                  // tensor_dim0_stride[31:0]
    g1[6] = 0;
    g1[7] = 0;
    i32x4 gz = {0, 0, 0, 0};
#if defined(__clang_major__) && (__clang_major__ >= 23)
    i32x8 gz8 = {0, 0, 0, 0, 0, 0, 0, 0};
    __builtin_amdgcn_tensor_load_to_lds(g0, g1, gz, gz, gz8, 0);
#else
    __builtin_amdgcn_tensor_load_to_lds(g0, g1, gz, gz, 0);
#endif
}

static __device__ __forceinline__ unsigned lds_offset_of(const void* p) {
    // flat shared pointer -> low 32 bits == LDS byte address (ISA 10.2 aperture)
    return (unsigned)(unsigned long long)p;
}

// ---------------------------------------------------------------------------
// GEMM core macro. Expanded inside a kernel that declares:
//   __shared__ bf16_t panel[2 * PANEL_ELEMS];
//   int wid, lm, lh;  v8f c[4] = {};
// Panel is indexed DIRECTLY (integer buffer offset) so LDS reads lower to
// ds_load_b128, not flat_load. Block = 8 waves = 128(M) x 64(N) tile.
// ---------------------------------------------------------------------------
#define GEMM_CORE(Xp, WTp, Ktot, Ntot, m0v, n0v)                                      \
  do {                                                                                 \
    const int _niter = (Ktot) / 64;                                                    \
    if (wid == 0)                                                                      \
      tdm_load_panel((WTp) + (long)(n0v) * (Ktot), lds_offset_of(&panel[0]),           \
                     (Ktot), (Ntot));                                                  \
    v16bf _a0 = load_a_frag((Xp) + (long)(m0v) * (Ktot), (Ktot), lm, 0, lh);           \
    v16bf _a1 = load_a_frag((Xp) + (long)(m0v) * (Ktot), (Ktot), lm, 32, lh);          \
    for (int _it = 0; _it < _niter; ++_it) {                                           \
      const int _boff = (_it & 1) * PANEL_ELEMS;                                       \
      if (wid == 0) __builtin_amdgcn_s_wait_tensorcnt(0);                              \
      __syncthreads(); /* buf[_boff] ready; other buffer drained */                    \
      if (wid == 0 && (_it + 1) < _niter)                                              \
        tdm_load_panel((WTp) + (long)(n0v) * (Ktot) + (_it + 1) * 64,                  \
                       lds_offset_of(&panel[PANEL_ELEMS - _boff]), (Ktot), (Ntot));    \
      v16bf _na0, _na1;                                                                \
      if ((_it + 1) < _niter) { /* prefetch next A while WMMAs run */                  \
        _na0 = load_a_frag((Xp) + (long)(m0v) * (Ktot), (Ktot), lm, (_it + 1) * 64, lh);      \
        _na1 = load_a_frag((Xp) + (long)(m0v) * (Ktot), (Ktot), lm, (_it + 1) * 64 + 32, lh); \
      }                                                                                \
      _Pragma("unroll")                                                                \
      for (int _nb = 0; _nb < 4; ++_nb) {                                              \
        const int _r = _boff + (_nb * 16 + lm) * PANEL_LDW + 16 * lh;                  \
        v8bf _l0 = *(const v8bf*)&panel[_r];                                           \
        v8bf _h0 = *(const v8bf*)&panel[_r + 8];                                       \
        v8bf _l1 = *(const v8bf*)&panel[_r + 32];                                      \
        v8bf _h1 = *(const v8bf*)&panel[_r + 40];                                      \
        v16bf _b0 = __builtin_shufflevector(_l0, _h0, 0, 1, 2, 3, 4, 5, 6, 7,          \
                                            8, 9, 10, 11, 12, 13, 14, 15);             \
        v16bf _b1 = __builtin_shufflevector(_l1, _h1, 0, 1, 2, 3, 4, 5, 6, 7,          \
                                            8, 9, 10, 11, 12, 13, 14, 15);             \
        c[_nb] = WMMA_BF16(_a0, _b0, c[_nb]);                                          \
        c[_nb] = WMMA_BF16(_a1, _b1, c[_nb]);                                          \
      }                                                                                \
      _a0 = _na0; _a1 = _na1;                                                          \
      __syncthreads(); /* all reads of this buffer done */                             \
    }                                                                                  \
  } while (0)

// ---------------------------------------------------------------------------
// Preprocessing kernels
// ---------------------------------------------------------------------------
__global__ void cvt_f32_to_bf16(const float* __restrict__ src, bf16_t* __restrict__ dst, int n) {
    for (int i = blockIdx.x * blockDim.x + threadIdx.x; i < n; i += gridDim.x * blockDim.x)
        dst[i] = (bf16_t)src[i];
}

// src: [K,N] fp32 row-major -> dst: [N,K] bf16 (transposed)
__global__ void transpose_to_bf16(const float* __restrict__ src, bf16_t* __restrict__ dst,
                                  int K, int N) {
    long total = (long)K * N;
    for (long i = (long)blockIdx.x * blockDim.x + threadIdx.x; i < total;
         i += (long)gridDim.x * blockDim.x) {
        long n = i / K, k = i % K;
        dst[i] = (bf16_t)src[k * (long)N + n];
    }
}

// ---------------------------------------------------------------------------
// QKV projection: X[8192,1024] @ Wqkv[1024,3072] + b.
// Scatters Q,K as [B,H,S,DH] and V transposed as [B,H,DH,S] (bf16).
// ---------------------------------------------------------------------------
__global__ __launch_bounds__(256) void qkv_gemm(const bf16_t* __restrict__ X,
                                                const bf16_t* __restrict__ WT,
                                                const float* __restrict__ bias,
                                                bf16_t* __restrict__ Q,
                                                bf16_t* __restrict__ Kh,
                                                bf16_t* __restrict__ VT) {
    __shared__ bf16_t panel[2 * PANEL_ELEMS];
    const int wid = threadIdx.x >> 5;
    const int lid = threadIdx.x & 31, lh = lid >> 4, lm = lid & 15;
    const int NT = NQKV_ / 64;
    const int mb = blockIdx.x / NT, nbk = blockIdx.x % NT;
    const int m0 = mb * 128 + wid * 16, n0 = nbk * 64;

    v8f c[4] = {};
    GEMM_CORE(X, WT, D_, NQKV_, m0, n0);

#pragma unroll
    for (int nb = 0; nb < 4; ++nb) {
        const int n = n0 + nb * 16 + lm;
        const float bv = bias[n];
        const int sec = n / D_;                 // 0=Q, 1=K, 2=V
        const int nn = n % D_;
        const int h = nn / DH_, d = nn % DH_;
#pragma unroll
        for (int i = 0; i < 8; ++i) {
            const int m = m0 + i + 8 * lh;      // C layout: row = vgpr + 8*half
            const int b = m / S_, s = m % S_;
            const bf16_t val = (bf16_t)(c[nb][i] + bv);
            if (sec == 0)      Q [(((long)(b * H_ + h) * S_ + s) * DH_) + d] = val;
            else if (sec == 1) Kh[(((long)(b * H_ + h) * S_ + s) * DH_) + d] = val;
            else               VT[(((long)(b * H_ + h) * DH_ + d) * S_) + s] = val;
        }
    }
}

// ---------------------------------------------------------------------------
// Flash attention: one wave owns a 16-query tile; streams 32-key blocks with
// online softmax; P restriped C-layout -> A-layout via per-wave LDS tile.
// ---------------------------------------------------------------------------
__global__ __launch_bounds__(128) void flash_attn(const bf16_t* __restrict__ Q,
                                                  const bf16_t* __restrict__ K,
                                                  const bf16_t* __restrict__ VT,
                                                  bf16_t* __restrict__ CTX) {
    __shared__ bf16_t plds[4][16 * 32];
    const int wid = threadIdx.x >> 5;
    const int lid = threadIdx.x & 31, lh = lid >> 4, lm = lid & 15;
    const int tile = blockIdx.x * 4 + wid;
    const int qt = tile % (S_ / 16);
    const int bh = tile / (S_ / 16);
    const int h = bh % H_, b = bh / H_;
    if (b >= B_) return;
    const int q0 = qt * 16;

    const bf16_t* Qh = Q + (long)(b * H_ + h) * S_ * DH_;
    const bf16_t* Kb = K + (long)(b * H_ + h) * S_ * DH_;
    const bf16_t* Vh = VT + (long)(b * H_ + h) * DH_ * S_;

    const v16bf aq0 = load_a_frag(Qh + (long)q0 * DH_, DH_, lm, 0, lh);
    const v16bf aq1 = load_a_frag(Qh + (long)q0 * DH_, DH_, lm, 32, lh);

    float mrun[8], lrun[8];
    v8f ctx[4] = {};
#pragma unroll
    for (int i = 0; i < 8; ++i) { mrun[i] = -1e30f; lrun[i] = 0.f; }

    const int kbmax = (q0 + 15) / 32;
    for (int kb = 0; kb <= kbmax; ++kb) {
        const int kk = kb * 32;
        v8f sc[2] = {};
#pragma unroll
        for (int j = 0; j < 2; ++j) {
            v16bf bk0 = load_b_frag(Kb + (long)(kk + j * 16) * DH_, DH_, lm, 0, lh);
            v16bf bk1 = load_b_frag(Kb + (long)(kk + j * 16) * DH_, DH_, lm, 32, lh);
            sc[j] = WMMA_BF16(aq0, bk0, sc[j]);
            sc[j] = WMMA_BF16(aq1, bk1, sc[j]);
        }
#pragma unroll
        for (int j = 0; j < 2; ++j)
#pragma unroll
            for (int i = 0; i < 8; ++i) {
                float s = sc[j][i] * 0.125f;     // 1/sqrt(64)
                const int row = q0 + i + 8 * lh;
                const int key = kk + j * 16 + lm;
                sc[j][i] = (key > row) ? -1e30f : s;
            }
#pragma unroll
        for (int i = 0; i < 8; ++i) {
            float rm = fmaxf(sc[0][i], sc[1][i]);
#pragma unroll
            for (int off = 8; off >= 1; off >>= 1) rm = fmaxf(rm, __shfl_xor(rm, off, 16));
            const float mnew = fmaxf(mrun[i], rm);
            const float p0 = __expf(sc[0][i] - mnew);
            const float p1 = __expf(sc[1][i] - mnew);
            float rs = p0 + p1;
#pragma unroll
            for (int off = 8; off >= 1; off >>= 1) rs += __shfl_xor(rs, off, 16);
            const float alpha = __expf(mrun[i] - mnew);
            lrun[i] = lrun[i] * alpha + rs;
            mrun[i] = mnew;
#pragma unroll
            for (int nb = 0; nb < 4; ++nb) ctx[nb][i] *= alpha;
            const int row = i + 8 * lh;
            plds[wid][row * 32 + lm]      = (bf16_t)p0;
            plds[wid][row * 32 + 16 + lm] = (bf16_t)p1;
        }
        asm volatile("s_wait_dscnt 0" ::: "memory");   // wave-local LDS restripe
        v8bf p0 = *reinterpret_cast<const v8bf*>(&plds[wid][lm * 32 + 16 * lh]);
        v8bf p1 = *reinterpret_cast<const v8bf*>(&plds[wid][lm * 32 + 16 * lh + 8]);
        v16bf pf = __builtin_shufflevector(p0, p1, 0, 1, 2, 3, 4, 5, 6, 7,
                                           8, 9, 10, 11, 12, 13, 14, 15);
        asm volatile("s_wait_dscnt 0" ::: "memory");
#pragma unroll
        for (int nb = 0; nb < 4; ++nb) {
            v16bf bv = load_b_frag(Vh, S_, nb * 16 + lm, kk, lh);
            ctx[nb] = WMMA_BF16(pf, bv, ctx[nb]);
        }
    }
#pragma unroll
    for (int nb = 0; nb < 4; ++nb)
#pragma unroll
        for (int i = 0; i < 8; ++i) {
            const int srow = q0 + i + 8 * lh;
            const int n = h * DH_ + nb * 16 + lm;
            CTX[((long)(b * S_) + srow) * D_ + n] = (bf16_t)(ctx[nb][i] / lrun[i]);
        }
}

// ---------------------------------------------------------------------------
// Output projection: CTX[8192,1024] @ Wout[1024,1024] + b -> fp32 d_out
// ---------------------------------------------------------------------------
__global__ __launch_bounds__(256) void out_proj(const bf16_t* __restrict__ CTX,
                                                const bf16_t* __restrict__ WT,
                                                const float* __restrict__ bias,
                                                float* __restrict__ out) {
    __shared__ bf16_t panel[2 * PANEL_ELEMS];
    const int wid = threadIdx.x >> 5;
    const int lid = threadIdx.x & 31, lh = lid >> 4, lm = lid & 15;
    const int NT = D_ / 64;
    const int mb = blockIdx.x / NT, nbk = blockIdx.x % NT;
    const int m0 = mb * 128 + wid * 16, n0 = nbk * 64;

    v8f c[4] = {};
    GEMM_CORE(CTX, WT, D_, D_, m0, n0);

#pragma unroll
    for (int nb = 0; nb < 4; ++nb) {
        const int n = n0 + nb * 16 + lm;
        const float bv = bias[n];
#pragma unroll
        for (int i = 0; i < 8; ++i) {
            const int m = m0 + i + 8 * lh;
            out[(long)m * D_ + n] = c[nb][i] + bv;
        }
    }
}

// ---------------------------------------------------------------------------
// Host-side launcher
// ---------------------------------------------------------------------------
extern "C" void kernel_launch(void* const* d_in, const int* in_sizes, int n_in,
                              void* d_out, int out_size, void* d_ws, size_t ws_size,
                              hipStream_t stream) {
    const float* x     = (const float*)d_in[0];   // [B,S,D]
    const float* w_qkv = (const float*)d_in[1];   // [D,3D]
    const float* b_qkv = (const float*)d_in[2];   // [3D]
    const float* w_out = (const float*)d_in[3];   // [D,D]
    const float* b_out = (const float*)d_in[4];   // [D]
    float* out = (float*)d_out;

    char* ws = (char*)d_ws;
    const size_t szXB   = (size_t)M_ * D_ * sizeof(bf16_t);            // 16 MiB
    const size_t szWqkv = (size_t)NQKV_ * D_ * sizeof(bf16_t);         //  6 MiB
    const size_t szWout = (size_t)D_ * D_ * sizeof(bf16_t);            //  2 MiB
    const size_t szHead = (size_t)B_ * H_ * S_ * DH_ * sizeof(bf16_t); // 16 MiB

    bf16_t* xb    = (bf16_t*)(ws);
    bf16_t* wqkvT = (bf16_t*)(ws + szXB);
    bf16_t* woutT = (bf16_t*)(ws + szXB + szWqkv);
    bf16_t* Qb    = (bf16_t*)(ws + szXB + szWqkv + szWout);
    bf16_t* Kb    = (bf16_t*)(ws + szXB + szWqkv + szWout + szHead);
    bf16_t* VTb   = (bf16_t*)(ws + szXB + szWqkv + szWout + 2 * szHead);
    bf16_t* ctxb  = (bf16_t*)(ws + szXB + szWqkv + szWout + 3 * szHead);

    cvt_f32_to_bf16<<<4096, 256, 0, stream>>>(x, xb, M_ * D_);
    transpose_to_bf16<<<4096, 256, 0, stream>>>(w_qkv, wqkvT, D_, NQKV_);
    transpose_to_bf16<<<2048, 256, 0, stream>>>(w_out, woutT, D_, D_);

    {   // QKV projection: 64 x 48 blocks of 128x64
        const int blocks = (M_ / 128) * (NQKV_ / 64);
        qkv_gemm<<<blocks, 256, 0, stream>>>(xb, wqkvT, b_qkv, Qb, Kb, VTb);
    }
    {   // flash attention: 8192 query tiles, 4 waves/block
        const int tiles = B_ * H_ * (S_ / 16);
        flash_attn<<<tiles / 4, 128, 0, stream>>>(Qb, Kb, VTb, ctxb);
    }
    {   // output projection: 64 x 16 blocks of 128x64
        const int blocks = (M_ / 128) * (D_ / 64);
        out_proj<<<blocks, 256, 0, stream>>>(ctxb, woutT, b_out, out);
    }
}
